// CAM_Module_33741263077583
// MI455X (gfx1250) — compile-verified, matrix operand order checked by reading, not run on hardware
//
#include <hip/hip_runtime.h>

// ---------------------------------------------------------------------------
// CAM module (channel attention) for MI455X / gfx1250, wave32, bf16 WMMA.
//   energy = A A^T  (A = [256 x 65536] fp32, per batch, row-contiguous)
//   attn   = softmax(rowmin - energy)      (algebraic form of the reference)
//   y      = gamma * (attn A) + x          (same flat layout, no transpose)
// Emulated-fp32 Gram via split bf16: E ~= hi*hi^T + hi*lo^T + lo*hi^T.
// Staging is software-pipelined across the LDS barrier; feats/output use
// non-temporal cache hints. Fragment tile offsets are compile-time so they
// fold into ds_load_b128 immediates (one address VGPR per LDS plane).
// ---------------------------------------------------------------------------

#define NPTS    65536
#define NCH     256
#define NBATCH  8
#define BATCH_ELEMS ((size_t)NPTS * NCH)
#define KSTR    40          // LDS stride in bf16 elems: mult of 8 (16B frag loads), conflict-free
#define SLICES  32          // K-slices for the Gram kernel
#define KCH     32          // K per bf16 WMMA step
#define TILEB   (16 * KSTR) // bf16 elems per 16-row tile step (1280 bytes)

typedef __attribute__((ext_vector_type(8)))  float        v8f;
typedef __attribute__((ext_vector_type(16))) __bf16       v16bf;
typedef __attribute__((ext_vector_type(4)))  float        f4v;
typedef __attribute__((ext_vector_type(4)))  unsigned int u4v;
typedef __attribute__((ext_vector_type(2)))  unsigned int u2v;

union Pack8 { __bf16 e[8]; u4v v; };
union Pack4 { __bf16 e[4]; u2v v; };
union Frag  { struct { u4v a, b; } u; v16bf v; };

__device__ __forceinline__ v8f wmma_bf16(v16bf a, v16bf b, v8f c) {
  // (neg_a, A, neg_b, B, c_mod, C, reuse_a, reuse_b)
  return __builtin_amdgcn_wmma_f32_16x16x32_bf16(false, a, false, b, (short)0, c,
                                                 false, false);
}

// CDNA5 16-bit operand layout (ISA 7.12.2): lane&15 = row index (M for A,
// N for B, assuming dense B mirrors A); lanes 0-15 hold K {0..7, 16..23},
// lanes 16-31 hold K {8..15, 24..31}, packed as consecutive halves.
// p must already include the per-lane offset; tile offsets fold to immediates.
__device__ __forceinline__ v16bf load_frag_at(const __bf16* __restrict p) {
  Frag f;
  f.u.a = *(const u4v*)(p);        // K + 0..7
  f.u.b = *(const u4v*)(p + 16);   // K + 16..23
  return f.v;
}

__device__ __forceinline__ int lane_off(int lane) {
  return (lane & 15) * KSTR + ((lane & 16) ? 8 : 0);
}

__global__ void zero_f32(float* __restrict p, int n) {
  int i = blockIdx.x * blockDim.x + threadIdx.x;
  if (i < n) p[i] = 0.0f;
}

// ---------------------------------------------------------------------------
// Phase 1: partial Gram per K-slice, atomically accumulated in fp32.
// grid = (SLICES, NBATCH), block = 1024 (32 waves). Each wave: 32x64 output
// block = 8 accum tiles, 24 v_wmma per K-chunk. Next chunk's global loads are
// issued right after the barrier so HBM latency overlaps the WMMA block.
// ---------------------------------------------------------------------------
__global__ __launch_bounds__(1024) void gram_kernel(const float* __restrict feats,
                                                    float* __restrict energy) {
  __shared__ __bf16 sh[NCH * KSTR];
  __shared__ __bf16 sl[NCH * KSTR];
  const int b = blockIdx.y;
  const float* __restrict A = feats + (size_t)b * BATCH_ELEMS;
  const int t = threadIdx.x, lane = t & 31, wave = t >> 5;
  const int wm = wave >> 2, wn = wave & 3;       // 8 x 4 wave grid over 256x256

  v8f acc[2][4];
  #pragma unroll
  for (int mt = 0; mt < 2; ++mt)
    #pragma unroll
    for (int nt = 0; nt < 4; ++nt) { v8f z = {}; acc[mt][nt] = z; }

  const int KPS  = NPTS / SLICES;
  const int k0   = blockIdx.x * KPS;
  const int srow = t >> 2;                        // 0..255
  const int skk  = (t & 3) * 8;                   // 0,8,16,24
  const float* __restrict src = A + (size_t)srow * NPTS + (size_t)(k0 + skk);

  // one per-lane LDS base per plane/role; tile steps become DS immediates
  const int lo = lane_off(lane);
  const __bf16* __restrict pAh = sh + wm * (2 * TILEB) + lo;
  const __bf16* __restrict pAl = sl + wm * (2 * TILEB) + lo;
  const __bf16* __restrict pBh = sh + wn * (4 * TILEB) + lo;
  const __bf16* __restrict pBl = sl + wn * (4 * TILEB) + lo;
  __bf16* __restrict sth = &sh[srow * KSTR + skk];
  __bf16* __restrict stl = &sl[srow * KSTR + skk];

  // prologue prefetch (chunk 0)
  f4v x0 = __builtin_nontemporal_load((const f4v*)(src));
  f4v x1 = __builtin_nontemporal_load((const f4v*)(src + 4));

  for (int kc = 0; kc < KPS; kc += KCH) {
    // ---- convert current chunk regs -> bf16 hi/lo, store to LDS ----
    Pack8 ph, pl;
    #pragma unroll
    for (int i = 0; i < 8; ++i) {
      float x = (i < 4) ? x0[i] : x1[i - 4];
      __bf16 h = (__bf16)x;
      ph.e[i] = h;
      pl.e[i] = (__bf16)(x - (float)h);
    }
    *(u4v*)(sth) = ph.v;
    *(u4v*)(stl) = pl.v;
    __syncthreads();

    // ---- prefetch next chunk (pointer step; re-load last chunk at end) ----
    src += (kc + KCH < KPS) ? KCH : 0;
    f4v nx0 = __builtin_nontemporal_load((const f4v*)(src));
    f4v nx1 = __builtin_nontemporal_load((const f4v*)(src + 4));

    // ---- compute: 3-product split-bf16 Gram tiles ----
    v16bf ah[2], al[2];
    #pragma unroll
    for (int mt = 0; mt < 2; ++mt) {
      ah[mt] = load_frag_at(pAh + mt * TILEB);
      al[mt] = load_frag_at(pAl + mt * TILEB);
    }
    #pragma unroll
    for (int nt = 0; nt < 4; ++nt) {
      v16bf bh = load_frag_at(pBh + nt * TILEB);
      v16bf bl = load_frag_at(pBl + nt * TILEB);
      #pragma unroll
      for (int mt = 0; mt < 2; ++mt) {
        acc[mt][nt] = wmma_bf16(ah[mt], bh, acc[mt][nt]);
        acc[mt][nt] = wmma_bf16(ah[mt], bl, acc[mt][nt]);
        acc[mt][nt] = wmma_bf16(al[mt], bh, acc[mt][nt]);
      }
    }
    __syncthreads();
    x0 = nx0;
    x1 = nx1;
  }

  // ---- accumulate partial Gram into global energy ----
  float* eb = energy + (size_t)b * (NCH * NCH);
  #pragma unroll
  for (int mt = 0; mt < 2; ++mt)
    #pragma unroll
    for (int nt = 0; nt < 4; ++nt)
      #pragma unroll
      for (int r = 0; r < 8; ++r) {
        int M = wm * 32 + mt * 16 + r + ((lane >> 4) << 3);  // C/D: vgpr r, lanes>=16 -> M+8
        int N = wn * 64 + nt * 16 + (lane & 15);
        __hip_atomic_fetch_add(&eb[M * NCH + N], acc[mt][nt][r],
                               __ATOMIC_RELAXED, __HIP_MEMORY_SCOPE_AGENT);
      }
}

// ---------------------------------------------------------------------------
// Phase 2: attn[c,d] = exp(rowmin - e[c,d]) / sum  (== softmax(rowmax - e))
// grid = (256 rows, NBATCH), block = 256. Output bf16 plane for GEMM2 A-op.
// ---------------------------------------------------------------------------
__global__ __launch_bounds__(256) void softmax_kernel(const float* __restrict energy,
                                                      __bf16* __restrict attn) {
  const int c = blockIdx.x, b = blockIdx.y, d = threadIdx.x;
  __shared__ float red[256];
  const size_t base = (size_t)b * (NCH * NCH) + (size_t)c * NCH;
  float e = energy[base + d];

  red[d] = e; __syncthreads();
  #pragma unroll
  for (int s = 128; s > 0; s >>= 1) {
    if (d < s) red[d] = fminf(red[d], red[d + s]);
    __syncthreads();
  }
  float mn = red[0]; __syncthreads();

  float w = __expf(mn - e);
  red[d] = w; __syncthreads();
  #pragma unroll
  for (int s = 128; s > 0; s >>= 1) {
    if (d < s) red[d] += red[d + s];
    __syncthreads();
  }
  attn[base + d] = (__bf16)(w / red[0]);
}

// ---------------------------------------------------------------------------
// Phase 3: y = gamma * (attn @ A) + x.
// grid = (NPTS/256 slabs, NBATCH), block = 1024 (32 waves).
// Feats K-chunk is transposed in registers -> LDS as [j][d] so the B fragment
// uses the same K-contiguous ds_load_b128 loader. attn hi x (feats hi + lo).
// Staging prefetch is pipelined across the barrier like the Gram kernel.
// ---------------------------------------------------------------------------
__global__ __launch_bounds__(1024) void out_kernel(const float* __restrict feats,
                                                   const __bf16* __restrict attn,
                                                   const float* __restrict gamma_p,
                                                   float* __restrict out) {
  __shared__ __bf16 sa [NCH * KSTR];   // attn chunk, [c][k]
  __shared__ __bf16 sbh[256 * KSTR];   // feats chunk hi, transposed [j][d]
  __shared__ __bf16 sbl[256 * KSTR];   // feats chunk lo, transposed [j][d]
  const int b = blockIdx.y;
  const int j0 = blockIdx.x * 256;
  const float*  __restrict F  = feats + (size_t)b * BATCH_ELEMS;
  const __bf16* __restrict At = attn  + (size_t)b * (NCH * NCH);
  const float gamma = gamma_p[0];
  const int t = threadIdx.x, lane = t & 31, wave = t >> 5;
  const int wm = wave >> 2, wn = wave & 3;
  const bool isB = (t < 512);          // wave-uniform split of staging duty

  // staging geometry
  const int d4 = (t >> 6) * 4;         // B stagers: rows d4..d4+3 (0,4,...,28)
  const int jj = (t & 63) * 4;         // B stagers: cols jj..jj+3
  const int t2  = t - 512;             // A stagers
  const int row = t2 >> 1;
  const int kh  = (t2 & 1) * 16;

  const int lo = lane_off(lane);
  const __bf16* __restrict pA  = sa  + wm * (2 * TILEB) + lo;
  const __bf16* __restrict pBh = sbh + wn * (4 * TILEB) + lo;
  const __bf16* __restrict pBl = sbl + wn * (4 * TILEB) + lo;

  v8f acc[2][4];
  #pragma unroll
  for (int mt = 0; mt < 2; ++mt)
    #pragma unroll
    for (int nt = 0; nt < 4; ++nt) { v8f z = {}; acc[mt][nt] = z; }

  // prologue prefetch (chunk d0 = 0)
  const float*  __restrict srcB = F + (size_t)d4 * NPTS + (size_t)(j0 + jj);
  const __bf16* __restrict srcA = At + row * NCH + kh;
  f4v rB[4];
  u4v rA[2];
  if (isB) {
    #pragma unroll
    for (int rr = 0; rr < 4; ++rr)
      rB[rr] = __builtin_nontemporal_load((const f4v*)(srcB + (size_t)rr * NPTS));
  } else {
    rA[0] = *(const u4v*)(srcA);
    rA[1] = *(const u4v*)(srcA + 8);
  }

  for (int d0 = 0; d0 < NCH; d0 += KCH) {
    // ---- store current chunk to LDS ----
    if (isB) {
      #pragma unroll
      for (int cx = 0; cx < 4; ++cx) {
        Pack4 ph, pl;
        #pragma unroll
        for (int rr = 0; rr < 4; ++rr) {
          float x = rB[rr][cx];
          __bf16 h = (__bf16)x;
          ph.e[rr] = h;
          pl.e[rr] = (__bf16)(x - (float)h);
        }
        *(u2v*)(&sbh[(jj + cx) * KSTR + d4]) = ph.v;
        *(u2v*)(&sbl[(jj + cx) * KSTR + d4]) = pl.v;
      }
    } else {
      *(u4v*)(&sa[row * KSTR + kh])     = rA[0];
      *(u4v*)(&sa[row * KSTR + kh + 8]) = rA[1];
    }
    __syncthreads();

    // ---- prefetch next chunk (pointer step; re-load last chunk at end) ----
    const int step = (d0 + KCH < NCH) ? KCH : 0;
    if (isB) {
      srcB += (size_t)step * NPTS;
      #pragma unroll
      for (int rr = 0; rr < 4; ++rr)
        rB[rr] = __builtin_nontemporal_load((const f4v*)(srcB + (size_t)rr * NPTS));
    } else {
      srcA += step;
      rA[0] = *(const u4v*)(srcA);
      rA[1] = *(const u4v*)(srcA + 8);
    }

    // ---- compute: attn_hi x (feats_hi + feats_lo) ----
    v16bf ah[2];
    #pragma unroll
    for (int mt = 0; mt < 2; ++mt)
      ah[mt] = load_frag_at(pA + mt * TILEB);
    #pragma unroll
    for (int nt = 0; nt < 4; ++nt) {
      v16bf bh = load_frag_at(pBh + nt * TILEB);
      v16bf bl = load_frag_at(pBl + nt * TILEB);
      #pragma unroll
      for (int mt = 0; mt < 2; ++mt) {
        acc[mt][nt] = wmma_bf16(ah[mt], bh, acc[mt][nt]);
        acc[mt][nt] = wmma_bf16(ah[mt], bl, acc[mt][nt]);
      }
    }
    __syncthreads();
  }

  // ---- epilogue: y = gamma*out + x, flat [C, n] order within the batch ----
  float* __restrict Ob = out + (size_t)b * BATCH_ELEMS;
  #pragma unroll
  for (int mt = 0; mt < 2; ++mt)
    #pragma unroll
    for (int nt = 0; nt < 4; ++nt)
      #pragma unroll
      for (int r = 0; r < 8; ++r) {
        int c = wm * 32 + mt * 16 + r + ((lane >> 4) << 3);
        int j = j0 + wn * 64 + nt * 16 + (lane & 15);
        size_t idx = (size_t)c * NPTS + j;
        float x = __builtin_nontemporal_load(&F[idx]);
        __builtin_nontemporal_store(gamma * acc[mt][nt][r] + x, &Ob[idx]);
      }
}

// ---------------------------------------------------------------------------
extern "C" void kernel_launch(void* const* d_in, const int* in_sizes, int n_in,
                              void* d_out, int out_size, void* d_ws, size_t ws_size,
                              hipStream_t stream) {
  const float* feats = (const float*)d_in[0];
  const float* gamma = (const float*)d_in[1];
  float* outp = (float*)d_out;

  // workspace: [0, 2MB) energy fp32 8x256x256 ; [2MB, 3MB) attn bf16 8x256x256
  float*  energy = (float*)d_ws;
  __bf16* attn   = (__bf16*)((char*)d_ws + (size_t)NBATCH * NCH * NCH * sizeof(float));

  const int e_n = NBATCH * NCH * NCH;
  zero_f32<<<(e_n + 255) / 256, 256, 0, stream>>>(energy, e_n);

  dim3 g1(SLICES, NBATCH);
  gram_kernel<<<g1, 1024, 0, stream>>>(feats, energy);

  dim3 g2(NCH, NBATCH);
  softmax_kernel<<<g2, 256, 0, stream>>>(energy, attn);

  dim3 g3(NPTS / 256, NBATCH);
  out_kernel<<<g3, 1024, 0, stream>>>(feats, attn, gamma, outp);
}
